// CXLoss_11355893530870
// MI455X (gfx1250) — compile-verified, matrix operand order checked by reading, not run on hardware
//
#include <hip/hip_runtime.h>
#include <hip/hip_bf16.h>
#include <math.h>

typedef _Float16 f16;
typedef unsigned int u32;
typedef __attribute__((ext_vector_type(16))) _Float16 v16h;
typedef __attribute__((ext_vector_type(8)))  _Float16 v8h;
typedef __attribute__((ext_vector_type(8)))  float    v8f;
typedef __attribute__((ext_vector_type(4)))  float    v4f;
typedef __attribute__((ext_vector_type(4)))  unsigned int v4u;
typedef __attribute__((ext_vector_type(4)))  int v4i;
typedef __attribute__((ext_vector_type(8)))  int v8i;

namespace {
constexpr int   Nn = 2, Cc = 256, Pp = 4096;     // N, C, P=H*W (H=W=64)
constexpr int   NT = Pp / 16;                    // 256 row-tiles per matrix
constexpr int   LROW = 264;                      // LDS row stride in halfs (512B + 16B pad)
constexpr float SIGMA_INV = 10.0f;               // 1/0.1
constexpr float Bb  = 1.0f;
constexpr float EPS = 1e-5f;
constexpr float INV_G = 1.0f / 65.0f;            // 1/(H+1) == 1/(W+1)
}

__device__ __forceinline__ void sched_fence() {
#if __has_builtin(__builtin_amdgcn_sched_barrier)
  __builtin_amdgcn_sched_barrier(0);   // keep ds_load clause ahead of WMMA block
#endif
}

// ---------------------------------------------------------------------------
// TDM: DMA one 16x256-half tile (8KB) from global into LDS, padding 16B after
// every 512B row (D# pad feature) so LDS rows stride 528B -> conflict-free
// ds_load_b128 fragment reads.  Issued by one wave; tracked on TENSORcnt.
// ---------------------------------------------------------------------------
__device__ __forceinline__ void tdm_load_tile(u32 lds_off, const f16* gsrc) {
  unsigned long long ga = (unsigned long long)(uintptr_t)gsrc;
  v4u g0;
  g0[0] = 1u;                                           // count=1 (valid), user mode
  g0[1] = lds_off;                                      // LDS byte address
  g0[2] = (u32)(ga & 0xffffffffu);                      // global_addr[31:0]
  g0[3] = (u32)((ga >> 32) & 0x01ffffffu) | (2u << 30); // global_addr[56:32] | type=2
  v8i g1;
  g1[0] = (1 << 16)            // data_size = 2 bytes
        | (1 << 20)            // pad_enable
        | (6 << 22)            // pad_interval: 128 DWORDs (512B)
        | (3 << 25);           // pad_amount: 4 DWORDs (16B)
  g1[1] = (int)(256u << 16);   // tensor_dim0[15:0]=256 in bits[63:48]
  g1[2] = (int)(16u  << 16);   // tensor_dim0[31:16]=0, tensor_dim1[15:0]=16
  g1[3] = (int)(256u << 16);   // tensor_dim1[31:16]=0, tile_dim0=256
  g1[4] = 16;                  // tile_dim1=16, tile_dim2=0
  g1[5] = 256;                 // tensor_dim0_stride[31:0]=256
  g1[6] = 0;
  g1[7] = 0;
  v4i z4 = {0, 0, 0, 0};
#if __clang_major__ >= 23
  v8i z8 = {0, 0, 0, 0, 0, 0, 0, 0};
  __builtin_amdgcn_tensor_load_to_lds(g0, g1, z4, z4, z8, 0);
#else
  __builtin_amdgcn_tensor_load_to_lds(g0, g1, z4, z4, 0);
#endif
}

// ---------------------------------------------------------------------------
// Fragment loaders for v_wmma_f32_16x16x32_f16 (ISA 7.12.2): lane l holds
// outer index (l&15), K-runs {8h..8h+7} and {16+8h..23+8h}, h = l>>4.
// ---------------------------------------------------------------------------
__device__ __forceinline__ v16h load_frag(const f16* __restrict__ pm,
                                          int row0, int cblk, int lane) {
  const int half = (lane >> 4) & 1;
  const int idx  = lane & 15;
  const f16* r = pm + ((size_t)(row0 + idx) * Cc) + cblk + (half << 3);
  v8h lo = *(const v8h*)(r);
  v8h hi = *(const v8h*)(r + 16);
  v16h out;
#pragma unroll
  for (int i = 0; i < 8; ++i) { out[i] = lo[i]; out[i + 8] = hi[i]; }
  return out;
}

__device__ __forceinline__ v16h lds_frag(const f16* tile, int cblk, int lane) {
  const int half = (lane >> 4) & 1;
  const int idx  = lane & 15;
  const f16* r = tile + idx * LROW + cblk + (half << 3);
  v8h lo = *(const v8h*)(r);
  v8h hi = *(const v8h*)(r + 16);
  v16h out;
#pragma unroll
  for (int i = 0; i < 8; ++i) { out[i] = lo[i]; out[i + 8] = hi[i]; }
  return out;
}

// ---- K0: meanT[c] over (N,H,W) --------------------------------------------
__global__ __launch_bounds__(256) void kmean(const float* __restrict__ T,
                                             float* __restrict__ meanT) {
  const int c = blockIdx.x;
  float s = 0.f;
  for (int i = threadIdx.x; i < Nn * Pp; i += 256) {
    const int n = i >> 12, p = i & (Pp - 1);
    s += T[((size_t)n * Cc + c) * Pp + p];
  }
  __shared__ float red[256];
  red[threadIdx.x] = s; __syncthreads();
  for (int st = 128; st > 0; st >>= 1) {
    if (threadIdx.x < st) red[threadIdx.x] += red[threadIdx.x + st];
    __syncthreads();
  }
  if (threadIdx.x == 0) meanT[c] = red[0] / (float)(Nn * Pp);
}

// ---- K1: center + L2-normalize per patch, write f16 patch-major -----------
__global__ __launch_bounds__(256) void knorm(const float* __restrict__ I,
                                             const float* __restrict__ T,
                                             const float* __restrict__ meanT,
                                             f16* __restrict__ Ipm,
                                             f16* __restrict__ Tpm) {
  const int b = blockIdx.x;
  const int n = b >> 12, p = b & (Pp - 1);
  const int c = threadIdx.x;
  const float m  = meanT[c];
  const float iv = I[((size_t)n * Cc + c) * Pp + p] - m;
  const float tv = T[((size_t)n * Cc + c) * Pp + p] - m;
  __shared__ float ri[256], rt[256];
  ri[c] = iv * iv; rt[c] = tv * tv; __syncthreads();
  for (int st = 128; st > 0; st >>= 1) {
    if (c < st) { ri[c] += ri[c + st]; rt[c] += rt[c + st]; }
    __syncthreads();
  }
  const float invI = 1.0f / sqrtf(ri[0]);
  const float invT = 1.0f / sqrtf(rt[0]);
  Ipm[((size_t)n * Pp + p) * Cc + c] = (f16)(iv * invI);
  Tpm[((size_t)n * Pp + p) * Cc + c] = (f16)(tv * invT);
}

// ---- K2: mn[n][p] = (1 - max_q dot)/2  (WMMA pass 1, TDM-staged A) --------
__global__ __launch_bounds__(256) void kcolmax(const f16* __restrict__ Ipm,
                                               const f16* __restrict__ Tpm,
                                               float* __restrict__ mn) {
  __shared__ alignas(16) f16 tile[2][16 * LROW];
  const int lane = threadIdx.x & 31;
  const int wv   = threadIdx.x >> 5;
  const int n = blockIdx.x >> 5;
  const int ptile = ((blockIdx.x & 31) * 8 + wv) << 4;
  const f16* Ib = Ipm + (size_t)n * Pp * Cc;
  const f16* Tb = Tpm + (size_t)n * Pp * Cc;
  v16h Bf[8];
#pragma unroll
  for (int kb = 0; kb < 8; ++kb) Bf[kb] = load_frag(Ib, ptile, kb * 32, lane);
  if (threadIdx.x < 32)
    tdm_load_tile((u32)(uintptr_t)&tile[0][0], Tb);
  float cmax = -1e30f;
  for (int t = 0; t < NT; ++t) {
    __syncthreads();                       // buf[(t+1)&1] free for overwrite
    if (threadIdx.x < 32) {
      if (t + 1 < NT) {
        tdm_load_tile((u32)(uintptr_t)&tile[(t + 1) & 1][0],
                      Tb + (size_t)(t + 1) * 16 * Cc);
        __builtin_amdgcn_s_wait_tensorcnt(1);   // tile t landed
      } else {
        __builtin_amdgcn_s_wait_tensorcnt(0);
      }
    }
    __syncthreads();                       // tile t visible to all waves
    const f16* tb = &tile[t & 1][0];
    v16h Af[8];
#pragma unroll
    for (int kb = 0; kb < 8; ++kb) Af[kb] = lds_frag(tb, kb * 32, lane);
    sched_fence();
    v8f acc0 = {}, acc1 = {};
#pragma unroll
    for (int kb = 0; kb < 8; kb += 2) {
      acc0 = __builtin_amdgcn_wmma_f32_16x16x32_f16(false, Af[kb],     false, Bf[kb],
                                                    (short)0, acc0, false, false);
      acc1 = __builtin_amdgcn_wmma_f32_16x16x32_f16(false, Af[kb + 1], false, Bf[kb + 1],
                                                    (short)0, acc1, false, false);
    }
#pragma unroll
    for (int r = 0; r < 8; ++r) cmax = fmaxf(cmax, acc0[r] + acc1[r]);
  }
  cmax = fmaxf(cmax, __shfl_xor(cmax, 16, 32));
  if (lane < 16) mn[(size_t)n * Pp + ptile + lane] = (1.0f - cmax) * 0.5f;
}

// ---- K3: s[n][p] = sum_q exp((B - raw/(mn+eps))/sigma)  (WMMA pass 2) -----
__global__ __launch_bounds__(256) void kcolsum(const f16* __restrict__ Ipm,
                                               const f16* __restrict__ Tpm,
                                               const float* __restrict__ mn,
                                               float* __restrict__ s) {
  __shared__ alignas(16) f16 tile[2][16 * LROW];
  const int lane = threadIdx.x & 31;
  const int wv   = threadIdx.x >> 5;
  const int n = blockIdx.x >> 5;
  const int ptile = ((blockIdx.x & 31) * 8 + wv) << 4;
  const f16* Ib = Ipm + (size_t)n * Pp * Cc;
  const f16* Tb = Tpm + (size_t)n * Pp * Cc;
  v16h Bf[8];
#pragma unroll
  for (int kb = 0; kb < 8; ++kb) Bf[kb] = load_frag(Ib, ptile, kb * 32, lane);
  const int p = ptile + (lane & 15);
  const float inv_mn = 1.0f / (mn[(size_t)n * Pp + p] + EPS);
  if (threadIdx.x < 32)
    tdm_load_tile((u32)(uintptr_t)&tile[0][0], Tb);
  float ssum = 0.f;
  for (int t = 0; t < NT; ++t) {
    __syncthreads();
    if (threadIdx.x < 32) {
      if (t + 1 < NT) {
        tdm_load_tile((u32)(uintptr_t)&tile[(t + 1) & 1][0],
                      Tb + (size_t)(t + 1) * 16 * Cc);
        __builtin_amdgcn_s_wait_tensorcnt(1);
      } else {
        __builtin_amdgcn_s_wait_tensorcnt(0);
      }
    }
    __syncthreads();
    const f16* tb = &tile[t & 1][0];
    v16h Af[8];
#pragma unroll
    for (int kb = 0; kb < 8; ++kb) Af[kb] = lds_frag(tb, kb * 32, lane);
    sched_fence();
    v8f acc0 = {}, acc1 = {};
#pragma unroll
    for (int kb = 0; kb < 8; kb += 2) {
      acc0 = __builtin_amdgcn_wmma_f32_16x16x32_f16(false, Af[kb],     false, Bf[kb],
                                                    (short)0, acc0, false, false);
      acc1 = __builtin_amdgcn_wmma_f32_16x16x32_f16(false, Af[kb + 1], false, Bf[kb + 1],
                                                    (short)0, acc1, false, false);
    }
#pragma unroll
    for (int r = 0; r < 8; ++r) {
      const float raw = (1.0f - (acc0[r] + acc1[r])) * 0.5f;
      ssum += __expf((Bb - raw * inv_mn) * SIGMA_INV);
    }
  }
  ssum += __shfl_xor(ssum, 16, 32);
  if (lane < 16) s[(size_t)n * Pp + ptile + lane] = ssum;
}

// ---- K4: spatial branch column stats mnsp[p], ssp[p] ----------------------
__global__ __launch_bounds__(256) void kspatial(float* __restrict__ mnsp,
                                                float* __restrict__ ssp) {
  const int p = blockIdx.x * 256 + threadIdx.x;
  const float gr = (float)(p >> 6) * INV_G, gc = (float)(p & 63) * INV_G;
  const float sqp = gr * gr + gc * gc;
  float mnv = 1e30f;
  for (int q = 0; q < Pp; ++q) {
    const float qr = (float)(q >> 6) * INV_G, qc = (float)(q & 63) * INV_G;
    const float d = fmaxf(sqp + qr * qr + qc * qc - 2.f * (gr * qr + gc * qc), 0.f);
    mnv = fminf(mnv, d);
  }
  const float inv = 1.0f / (mnv + EPS);
  float ss = 0.f;
  for (int q = 0; q < Pp; ++q) {
    const float qr = (float)(q >> 6) * INV_G, qc = (float)(q & 63) * INV_G;
    const float d = fmaxf(sqp + qr * qr + qc * qc - 2.f * (gr * qr + gc * qc), 0.f);
    ss += __expf((Bb - d * inv) * SIGMA_INV);
  }
  mnsp[p] = mnv; ssp[p] = ss;
}

// ---- K4b: fold per-column stats into one float4 table ---------------------
__global__ __launch_bounds__(256) void ktbl(const float* __restrict__ mn,
                                            const float* __restrict__ s,
                                            const float* __restrict__ mnsp,
                                            const float* __restrict__ ssp,
                                            v4f* __restrict__ tbl) {
  const int i = blockIdx.x * 256 + threadIdx.x;   // i < N*P
  const int p = i & (Pp - 1);
  v4f t;
  t.x = 1.0f / (mn[i] + EPS);
  t.y = 0.5f / s[i];
  t.z = 1.0f / (mnsp[p] + EPS);
  t.w = 0.5f / ssp[p];
  tbl[i] = t;
}

// ---- K5: m[n][q] = max_p (0.5*cx_feat + 0.5*cx_sp)  (WMMA pass 3) ---------
__global__ __launch_bounds__(256) void krowmax(const f16* __restrict__ Ipm,
                                               const f16* __restrict__ Tpm,
                                               const v4f* __restrict__ tbl,
                                               float* __restrict__ m) {
  __shared__ alignas(16) f16 tile[2][16 * LROW];
  const int lane = threadIdx.x & 31;
  const int wv   = threadIdx.x >> 5;
  const int n = blockIdx.x >> 5;
  const int qtile = ((blockIdx.x & 31) * 8 + wv) << 4;
  const f16* Ib = Ipm + (size_t)n * Pp * Cc;
  const f16* Tb = Tpm + (size_t)n * Pp * Cc;
  v16h Bf[8];                                     // held: B columns = q (T patches)
#pragma unroll
  for (int kb = 0; kb < 8; ++kb) Bf[kb] = load_frag(Tb, qtile, kb * 32, lane);
  const int q  = qtile + (lane & 15);
  const float qr = (float)(q >> 6) * INV_G, qc = (float)(q & 63) * INV_G;
  const float sqq = qr * qr + qc * qc;
  const int mrow = (lane >> 4) << 3;              // M offset for this lane half
  if (threadIdx.x < 32)
    tdm_load_tile((u32)(uintptr_t)&tile[0][0], Ib);
  float mmax = -1e30f;
  for (int t = 0; t < NT; ++t) {
    __syncthreads();
    if (threadIdx.x < 32) {
      if (t + 1 < NT) {
        tdm_load_tile((u32)(uintptr_t)&tile[(t + 1) & 1][0],
                      Ib + (size_t)(t + 1) * 16 * Cc);
        __builtin_amdgcn_s_wait_tensorcnt(1);
      } else {
        __builtin_amdgcn_s_wait_tensorcnt(0);
      }
    }
    __syncthreads();
    const f16* tb = &tile[t & 1][0];
    v16h Af[8];
#pragma unroll
    for (int kb = 0; kb < 8; ++kb) Af[kb] = lds_frag(tb, kb * 32, lane);
    sched_fence();
    v8f acc0 = {}, acc1 = {};
#pragma unroll
    for (int kb = 0; kb < 8; kb += 2) {
      acc0 = __builtin_amdgcn_wmma_f32_16x16x32_f16(false, Af[kb],     false, Bf[kb],
                                                    (short)0, acc0, false, false);
      acc1 = __builtin_amdgcn_wmma_f32_16x16x32_f16(false, Af[kb + 1], false, Bf[kb + 1],
                                                    (short)0, acc1, false, false);
    }
    const int pt = t << 4;
#pragma unroll
    for (int r = 0; r < 8; ++r) {
      const int p = pt + r + mrow;
      const v4f tc = tbl[(size_t)n * Pp + p];
      const float raw = (1.0f - (acc0[r] + acc1[r])) * 0.5f;
      const float wf  = __expf((Bb - raw * tc.x) * SIGMA_INV) * tc.y;
      const float pr = (float)(p >> 6) * INV_G, pc = (float)(p & 63) * INV_G;
      const float d  = fmaxf(pr * pr + pc * pc + sqq - 2.f * (pr * qr + pc * qc), 0.f);
      const float wsp = __expf((Bb - d * tc.z) * SIGMA_INV) * tc.w;
      mmax = fmaxf(mmax, wf + wsp);
    }
  }
  mmax = fmaxf(mmax, __shfl_xor(mmax, 16, 32));
  if (lane < 16) m[(size_t)n * Pp + qtile + lane] = mmax;
}

// ---- K6: out = mean_n( -log( mean_q m[n,q] ) ) ----------------------------
__global__ __launch_bounds__(256) void kfinal(const float* __restrict__ m,
                                              float* __restrict__ out) {
  __shared__ float red[256];
  float acc = 0.f;
  for (int n = 0; n < Nn; ++n) {
    float ssum = 0.f;
    for (int qq = threadIdx.x; qq < Pp; qq += 256) ssum += m[(size_t)n * Pp + qq];
    red[threadIdx.x] = ssum; __syncthreads();
    for (int st = 128; st > 0; st >>= 1) {
      if (threadIdx.x < st) red[threadIdx.x] += red[threadIdx.x + st];
      __syncthreads();
    }
    if (threadIdx.x == 0) acc += -logf(red[0] / (float)Pp);
    __syncthreads();
  }
  if (threadIdx.x == 0) out[0] = acc / (float)Nn;
}

// ---------------------------------------------------------------------------
extern "C" void kernel_launch(void* const* d_in, const int* in_sizes, int n_in,
                              void* d_out, int out_size, void* d_ws, size_t ws_size,
                              hipStream_t stream) {
  (void)in_sizes; (void)n_in; (void)out_size; (void)ws_size;
  const float* I = (const float*)d_in[0];
  const float* T = (const float*)d_in[1];
  float* out = (float*)d_out;

  char* w = (char*)d_ws;
  const size_t MB = 1024 * 1024, KB = 1024;
  f16*   Ipm   = (f16*)(w + 0);                    // 4 MB  (N*P*C f16)
  f16*   Tpm   = (f16*)(w + 4 * MB);               // 4 MB
  float* meanT = (float*)(w + 8 * MB);             // 1 KB
  float* mn    = (float*)(w + 8 * MB + 4 * KB);    // 32 KB (N*P)
  float* s     = (float*)(w + 8 * MB + 64 * KB);   // 32 KB
  float* mnsp  = (float*)(w + 8 * MB + 128 * KB);  // 16 KB (P)
  float* ssp   = (float*)(w + 8 * MB + 160 * KB);  // 16 KB
  float* marr  = (float*)(w + 8 * MB + 192 * KB);  // 32 KB
  v4f*   tbl   = (v4f*)(w + 8 * MB + 256 * KB);    // 128 KB

  kmean  <<<Cc,       256, 0, stream>>>(T, meanT);
  knorm  <<<Nn * Pp,  256, 0, stream>>>(I, T, meanT, Ipm, Tpm);
  kcolmax<<<64,       256, 0, stream>>>(Ipm, Tpm, mn);          // 512 waves
  kcolsum<<<64,       256, 0, stream>>>(Ipm, Tpm, mn, s);
  kspatial<<<Pp / 256, 256, 0, stream>>>(mnsp, ssp);
  ktbl   <<<(Nn * Pp) / 256, 256, 0, stream>>>(mn, s, mnsp, ssp, tbl);
  krowmax<<<64,       256, 0, stream>>>(Ipm, Tpm, tbl, marr);
  kfinal <<<1,        256, 0, stream>>>(marr, out);
}
